// Gnn_24343874634231
// MI455X (gfx1250) — matrix operand order repaired and checked by reading, not compiled, not run on hardware
//
#include <hip/hip_runtime.h>
#include <hip/hip_bf16.h>

typedef __attribute__((ext_vector_type(16))) _Float16 v16h;
typedef __attribute__((ext_vector_type(8)))  float    v8f;

#if __has_builtin(__builtin_amdgcn_sched_group_barrier)
#define SCHED_GROUP(mask, size, id) __builtin_amdgcn_sched_group_barrier((mask), (size), (id))
#else
#define SCHED_GROUP(mask, size, id)
#endif

__device__ __forceinline__ float lrelu(float v) { return v > 0.0f ? v : 0.01f * v; }
__device__ __forceinline__ float4 lrelu4(float4 v) {
    v.x = lrelu(v.x); v.y = lrelu(v.y); v.z = lrelu(v.z); v.w = lrelu(v.w);
    return v;
}

// ---------------------------------------------------------------------------
// deg / dinv
// ---------------------------------------------------------------------------
__global__ void k_fill_one(float* __restrict__ deg, int n) {
    int i = blockIdx.x * 256 + threadIdx.x;
    if (i < n) deg[i] = 1.0f;                       // self-loop
}

__global__ void k_count_deg(const int* __restrict__ dst, int e, float* __restrict__ deg) {
    int i = blockIdx.x * 256 + threadIdx.x;
    if (i < e) atomicAdd(&deg[dst[i]], 1.0f);
}

__global__ void k_rsqrt_inplace(float* __restrict__ d, int n) {
    int i = blockIdx.x * 256 + threadIdx.x;
    if (i < n) d[i] = rsqrtf(d[i]);
}

// ---------------------------------------------------------------------------
// One-time weight prep: WT[n][k] = (f16) W[k][n]   (K x 128 -> 128 x K, f16)
// ---------------------------------------------------------------------------
__global__ void k_cvt_transpose_w(const float* __restrict__ W, _Float16* __restrict__ WT, int K) {
    int i = blockIdx.x * 256 + threadIdx.x;         // over K*128
    if (i < K * 128) {
        int k = i >> 7;
        int n = i & 127;
        WT[n * K + k] = (_Float16)W[i];
    }
}

// ---------------------------------------------------------------------------
// WMMA GEMM: out[N][128] = Acat[N][K] @ W[K][128]
//   Acat row = [A0 row (128) | A1 row (128)] when K==256, else A0 row.
//   Optional leaky-ReLU applied to A elements on load (layer-2 concat+lrelu).
//   f16 multiplicands, f32 accumulate via v_wmma_f32_16x16x32_f16.
// Block = 128 threads = 4 waves; wave -> 16 rows x 128 cols.
// WT is pre-converted transposed f16 weights (128 rows of K halves).
// ---------------------------------------------------------------------------
__global__ __launch_bounds__(128)
void k_gemm_wmma(const float* __restrict__ A0, const float* __restrict__ A1,
                 int applyLrelu, int N, int K,
                 const _Float16* __restrict__ WT, float* __restrict__ out)
{
    __shared__ _Float16 ldsW[256 * 128];            // WT[n][k], row stride K

    const int t = threadIdx.x;

    // Straight coalesced copy of pre-transposed f16 W into LDS (16B vectors).
    {
        const uint4* wsrc = (const uint4*)WT;
        uint4*       wdst = (uint4*)ldsW;
        const int nvec = K * 16;                    // (128*K halves)*2B / 16B
        for (int i = t; i < nvec; i += 128) wdst[i] = wsrc[i];
    }
    __syncthreads();

    const int wave    = t >> 5;
    const int lane    = t & 31;
    const int lo      = lane & 15;
    const int half    = lane >> 4;
    const int rowBase = blockIdx.x * 64 + wave * 16;
    const int arow    = min(rowBase + lo, N - 1);   // clamp: keeps EXEC uniform for WMMA

    const float* rowA0 = A0 + (size_t)arow * 128 + half * 16;
    const float* rowA1 = A1 + (size_t)arow * 128 + half * 16;

    v8f acc[8] = {};

    const int nK = K >> 5;                          // K-steps of 32
    for (int kt = 0; kt < nK; ++kt) {
        const int kbase = kt * 32;
        const float* src = (kbase < 128) ? (rowA0 + kbase) : (rowA1 + (kbase - 128));

        // A fragment, contiguous k order (same as B):
        //   e in [0,16): K = kbase + half*16 + e
        const float4* p = (const float4*)src;
        float4 f0 = p[0];
        float4 f1 = p[1];
        float4 g0 = p[2];
        float4 g1 = p[3];
        if (applyLrelu) {
            f0 = lrelu4(f0); f1 = lrelu4(f1); g0 = lrelu4(g0); g1 = lrelu4(g1);
        }
        v16h a;
        a[0]  = (_Float16)f0.x; a[1]  = (_Float16)f0.y; a[2]  = (_Float16)f0.z; a[3]  = (_Float16)f0.w;
        a[4]  = (_Float16)f1.x; a[5]  = (_Float16)f1.y; a[6]  = (_Float16)f1.z; a[7]  = (_Float16)f1.w;
        a[8]  = (_Float16)g0.x; a[9]  = (_Float16)g0.y; a[10] = (_Float16)g0.z; a[11] = (_Float16)g0.w;
        a[12] = (_Float16)g1.x; a[13] = (_Float16)g1.y; a[14] = (_Float16)g1.z; a[15] = (_Float16)g1.w;

        // Preload all 8 B fragments (16 ds_load_b128), then chain WMMAs.
        v16h b[8];
        #pragma unroll
        for (int nt = 0; nt < 8; ++nt)
            b[nt] = *(const v16h*)(&ldsW[(nt * 16 + lo) * K + kbase + half * 16]);

        #pragma unroll
        for (int nt = 0; nt < 8; ++nt)
            acc[nt] = __builtin_amdgcn_wmma_f32_16x16x32_f16(
                          false, a, false, b[nt], (short)0, acc[nt], false, false);

        // Pre-RA schedule shape per iteration:
        //   4x global_load_b128 (A tile) -> 16x ds_load (B tiles) -> 8x WMMA.
        // Keeping all DS reads ahead of the matrix group forces distinct B
        // registers, replacing s_wait_dscnt 0x0 stalls with staggered waits.
        SCHED_GROUP(0x020, 4, 0);   // VMEM reads
        SCHED_GROUP(0x100, 16, 0);  // DS reads
        SCHED_GROUP(0x008, 8, 0);   // matrix (WMMA) ops
    }

    // C/D layout: VGPR r holds row M = half*8 + r, col N = lane&15 (per 16-col tile).
    #pragma unroll
    for (int nt = 0; nt < 8; ++nt) {
        #pragma unroll
        for (int r = 0; r < 8; ++r) {
            int row = rowBase + half * 8 + r;
            if (row < N) out[(size_t)row * 128 + nt * 16 + lo] = acc[nt][r];
        }
    }
}

// ---------------------------------------------------------------------------
// Aggregation: agg = h*dinv^2 + bias  (self-loop + bias, also re-inits buffer)
// ---------------------------------------------------------------------------
__global__ void k_agg_init(const float* __restrict__ h, const float* __restrict__ dinv,
                           const float* __restrict__ bias, float* __restrict__ agg, int n128)
{
    int i = blockIdx.x * 256 + threadIdx.x;
    if (i < n128) {
        int row = i >> 7;
        int c   = i & 127;
        float dv = dinv[row];
        agg[i] = h[i] * dv * dv + bias[c];
    }
}

// One 32-lane group per edge: float4 gather of h[src], scaled atomic scatter to agg[dst].
__global__ void k_agg_edges(const int* __restrict__ ei, int e,
                            const float* __restrict__ h, const float* __restrict__ dinv,
                            float* __restrict__ agg)
{
    int edge = blockIdx.x * 8 + (threadIdx.x >> 5);
    int lane = threadIdx.x & 31;
    if (edge < e) {
        int s = ei[edge];
        int d = ei[e + edge];
        float nrm = dinv[s] * dinv[d];
        const float4* hp = (const float4*)(h + (size_t)s * 128);
        float4 v = hp[lane];
        float* ap = agg + (size_t)d * 128 + lane * 4;
        atomicAdd(ap + 0, v.x * nrm);
        atomicAdd(ap + 1, v.y * nrm);
        atomicAdd(ap + 2, v.z * nrm);
        atomicAdd(ap + 3, v.w * nrm);
    }
}

// ---------------------------------------------------------------------------
// out[i] = dot(lrelu(agg[i]), Wout) + bout   (32-lane reduction per node)
// ---------------------------------------------------------------------------
__global__ void k_final(const float* __restrict__ agg, const float* __restrict__ Wout,
                        const float* __restrict__ bout, float* __restrict__ out, int n)
{
    int node = blockIdx.x * 8 + (threadIdx.x >> 5);
    int lane = threadIdx.x & 31;
    if (node < n) {
        const float4* zp = (const float4*)(agg + (size_t)node * 128);
        float4 z = lrelu4(zp[lane]);
        float4 w = ((const float4*)Wout)[lane];
        float p = z.x * w.x + z.y * w.y + z.z * w.z + z.w * w.w;
        #pragma unroll
        for (int off = 16; off > 0; off >>= 1) p += __shfl_xor(p, off, 32);
        if (lane == 0) out[node] = p + bout[0];
    }
}

// ---------------------------------------------------------------------------
// Launch
// ---------------------------------------------------------------------------
extern "C" void kernel_launch(void* const* d_in, const int* in_sizes, int n_in,
                              void* d_out, int out_size, void* d_ws, size_t ws_size,
                              hipStream_t stream)
{
    const float* x    = (const float*)d_in[0];
    const int*   ei   = (const int*)  d_in[1];   // [2,E] flattened: src[0..E), dst[E..2E)
    const float* W1   = (const float*)d_in[2];
    const float* b1   = (const float*)d_in[3];
    const float* W2   = (const float*)d_in[4];
    const float* b2   = (const float*)d_in[5];
    const float* Wout = (const float*)d_in[6];
    const float* bout = (const float*)d_in[7];
    float* out = (float*)d_out;

    const int N = in_sizes[0] / 128;
    const int E = in_sizes[1] / 2;

    char* ws = (char*)d_ws;
    const size_t hBytes = (size_t)N * 128 * sizeof(float);
    float*     dinv = (float*)ws;                                          // N floats
    _Float16*  WT   = (_Float16*)(ws + (512u << 10));                      // up to 256x128 f16 (64KB)
    float*     h    = (float*)(ws + (1u << 20));                           // N x 128
    float*     agg  = (float*)(ws + (1u << 20) + ((hBytes + 255) & ~(size_t)255));

    const int n128 = N * 128;
    const dim3 blk(256);

    // degrees -> dinv
    k_fill_one      <<<(N + 255) / 256, blk, 0, stream>>>(dinv, N);
    k_count_deg     <<<(E + 255) / 256, blk, 0, stream>>>(ei + E, E, dinv);
    k_rsqrt_inplace <<<(N + 255) / 256, blk, 0, stream>>>(dinv, N);

    // layer 1: h = x @ W1 ; agg = D^-1/2 (A+I) D^-1/2 h + b1
    k_cvt_transpose_w <<<(128 * 128 + 255) / 256, blk, 0, stream>>>(W1, WT, 128);
    k_gemm_wmma <<<(N + 63) / 64, 128, 0, stream>>>(x, x, 0, N, 128, WT, h);
    k_agg_init  <<<(n128 + 255) / 256, blk, 0, stream>>>(h, dinv, b1, agg, n128);
    k_agg_edges <<<(E + 7) / 8, blk, 0, stream>>>(ei, E, h, dinv, agg);

    // layer 2: h = lrelu([x | agg]) @ W2 ; agg = normalize-aggregate + b2
    k_cvt_transpose_w <<<(256 * 128 + 255) / 256, blk, 0, stream>>>(W2, WT, 256);
    k_gemm_wmma <<<(N + 63) / 64, 128, 0, stream>>>(x, agg, 1, N, 256, WT, h);
    k_agg_init  <<<(n128 + 255) / 256, blk, 0, stream>>>(h, dinv, b2, agg, n128);
    k_agg_edges <<<(E + 7) / 8, blk, 0, stream>>>(ei, E, h, dinv, agg);

    // out = lrelu(agg) @ Wout + bout
    k_final <<<(N + 7) / 8, blk, 0, stream>>>(agg, Wout, bout, out, N);
}
